// PGExplainer_73581379715388
// MI455X (gfx1250) — matrix operand order repaired and checked by reading, not compiled, not run on hardware
//
#include <hip/hip_runtime.h>
#include <hip/hip_fp16.h>
#include <math.h>

typedef __attribute__((ext_vector_type(16))) _Float16 v16h;
typedef __attribute__((ext_vector_type(8)))  _Float16 v8h;
typedef __attribute__((ext_vector_type(8)))  float    v8f;
typedef __attribute__((ext_vector_type(4)))  float    v4f;

#define HIDDEN 128
#define DIN    272     // 2*HIDDEN + 16
#define NKB    9       // K blocks of 32 (272 padded to 288)
#define NNT    8       // N tiles of 16 (128 cols)
#define GEPS   1e-8f
#define BFRAG_BYTES (NKB * NNT * 512 * 2)   // 73,728 (256-aligned)

// -------------------------------------------------------------------------
// Prep: swizzle W1 (f32 [272][128], row = K) into f16 WMMA B-fragments.
// Fragment f = kk*8 + nt holds the 32x16 f16 B tile for K-block kk, N-tile nt.
// Per-lane layout (16-bit B, 32x16): N = lane%16;
//   lanes 0-15:  h[0..7]=K0..7,  h[8..15]=K16..23
//   lanes 16-31: h[0..7]=K8..15, h[8..15]=K24..31
// Storage: bfrag[f*512 + lane*16 + h] so a lane's v16h is one 32B load.
// -------------------------------------------------------------------------
__global__ void pgx_prep_w1(const float* __restrict__ W1,
                            _Float16* __restrict__ bfrag) {
    int t = blockIdx.x * blockDim.x + threadIdx.x;
    if (t >= NKB * NNT * 512) return;
    int f    = t >> 9;
    int pos  = t & 511;
    int lane = pos >> 4;
    int h    = pos & 15;
    int kk   = f >> 3;
    int nt   = f & 7;
    int n    = nt * 16 + (lane & 15);
    int kl   = (h < 8) ? ((lane >> 4) * 8 + h) : (16 + (lane >> 4) * 8 + (h - 8));
    int kg   = kk * 32 + kl;
    float v  = (kg < DIN) ? W1[kg * HIDDEN + n] : 0.0f;
    bfrag[t] = (_Float16)v;
}

// Prep: bulk f32 -> f16 conversion (for x and etype_emb).
__global__ void pgx_cvt_f16(const float* __restrict__ in,
                            _Float16* __restrict__ out, int n) {
    int i = blockIdx.x * blockDim.x + threadIdx.x;
    if (i < n) out[i] = (_Float16)in[i];
}

// --- 8-feature chunk loaders, producing f16 halves for the A fragment -----
__device__ __forceinline__ v8h pgx_load8h(const _Float16* __restrict__ p) {
    return *(const v8h*)p;                         // one 16B load
}
__device__ __forceinline__ v8h pgx_load8h(const float* __restrict__ p) {
    v4f q0 = ((const v4f*)p)[0];
    v4f q1 = ((const v4f*)p)[1];
    v8h r;
#pragma unroll
    for (int i = 0; i < 4; ++i) {
        r[i]     = (_Float16)q0[i];
        r[i + 4] = (_Float16)q1[i];
    }
    return r;
}

// 8 consecutive features starting at k (multiple of 8) of the virtual
// 288-wide row [ x[src](0..127) | x[dst](128..255) | emb(256..271) | 0 pad ].
template <typename T>
__device__ __forceinline__ v8h pgx_chunk(const T* __restrict__ rs,
                                         const T* __restrict__ rd,
                                         const T* __restrict__ re, int k) {
    if (k < 128)  return pgx_load8h(rs + k);
    if (k < 256)  return pgx_load8h(rd + (k - 128));
    if (k < DIN)  return pgx_load8h(re + (k - 256));
    v8h z = {(_Float16)0.0f, (_Float16)0.0f, (_Float16)0.0f, (_Float16)0.0f,
             (_Float16)0.0f, (_Float16)0.0f, (_Float16)0.0f, (_Float16)0.0f};
    return z;
}

// -------------------------------------------------------------------------
// Main kernel: one wave32 per 32 edges (2 M-tiles of 16).
// C/D tile layout (f32 16x16): lanes 0-15 -> N=lane, rows M=0..7 in slots 0..7;
// lanes 16-31 -> N=lane-16, rows M=8..15.
// A layout (f16 16x32): row M = lane%16;
//   lanes 0-15:  a[0..7]=K0..7,  a[8..15]=K16..23
//   lanes 16-31: a[0..7]=K8..15, a[8..15]=K24..31
// T = _Float16 (pre-converted gather tables in ws) or float (fallback).
// -------------------------------------------------------------------------
template <typename T>
__global__ void __launch_bounds__(256, 1)
pgx_main(const T*     __restrict__ xT,
         const int*   __restrict__ edge_index,
         const int*   __restrict__ edge_type,
         const float* __restrict__ u,
         const T*     __restrict__ embT,
         const _Float16* __restrict__ bfrag,
         const float* __restrict__ b1,
         const float* __restrict__ W2,
         const float* __restrict__ b2,
         float* __restrict__ out,
         int E) {
    const int lane = threadIdx.x & 31;
    const int wave = threadIdx.x >> 5;
    const int wid  = blockIdx.x * 8 + wave;
    const int e0   = wid * 32;
    if (e0 >= E) return;                  // wave-uniform: EXEC stays full

    const int mrow = lane & 15;
    const int hi   = lane >> 4;
    const int hio  = hi * 8;

    // Per-lane column data (N = nt*16 + mrow) for bias fold + second layer.
    float b1v[8], w2v[8];
#pragma unroll
    for (int nt = 0; nt < 8; ++nt) {
        b1v[nt] = b1[nt * 16 + mrow];
        w2v[nt] = W2[nt * 16 + mrow];
    }

    // Accumulators seeded with b1 (C-matrix of the WMMA).
    v8f c[2][8];
#pragma unroll
    for (int mt = 0; mt < 2; ++mt)
#pragma unroll
        for (int nt = 0; nt < 8; ++nt)
#pragma unroll
            for (int r = 0; r < 8; ++r)
                c[mt][nt][r] = b1v[nt];

    // Row gather pointers for this lane's row in each M-tile.
    const T* rs[2];
    const T* rd[2];
    const T* re[2];
#pragma unroll
    for (int mt = 0; mt < 2; ++mt) {
        int e = e0 + mt * 16 + mrow;
        if (e >= E) e = E - 1;            // clamp (E%32==0 in practice)
        int s  = edge_index[e];
        int d  = edge_index[E + e];
        int et = edge_type[e];
        rs[mt] = xT   + (size_t)s  * HIDDEN;
        rd[mt] = xT   + (size_t)d  * HIDDEN;
        re[mt] = embT + (size_t)et * 16;
    }

    // K loop: 9 blocks of 32; 16 WMMAs per block.
#pragma unroll
    for (int kk = 0; kk < NKB; ++kk) {
        v16h a[2];
#pragma unroll
        for (int mt = 0; mt < 2; ++mt) {
            const int c0 = kk * 32 + hio;        // -> a[0..7]
            const int c1 = kk * 32 + 16 + hio;   // -> a[8..15]
            v8h h0 = pgx_chunk(rs[mt], rd[mt], re[mt], c0);
            v8h h1 = pgx_chunk(rs[mt], rd[mt], re[mt], c1);
#pragma unroll
            for (int i = 0; i < 8; ++i) {
                a[mt][i]     = h0[i];
                a[mt][i + 8] = h1[i];
            }
        }
#pragma unroll
        for (int nt = 0; nt < 8; ++nt) {
            const v16h bw =
                *(const v16h*)(bfrag + ((size_t)(kk * 8 + nt) * 512 + lane * 16));
            c[0][nt] = __builtin_amdgcn_wmma_f32_16x16x32_f16(
                false, a[0], false, bw, (short)0, c[0][nt], false, false);
            c[1][nt] = __builtin_amdgcn_wmma_f32_16x16x32_f16(
                false, a[1], false, bw, (short)0, c[1][nt], false, false);
        }
    }

    // Epilogue: relu(h) . W2 — per-lane partial over its 8 columns, then
    // butterfly reduce across the 16-lane group.
    float part[2][8];
#pragma unroll
    for (int mt = 0; mt < 2; ++mt)
#pragma unroll
        for (int r = 0; r < 8; ++r) {
            float acc = 0.0f;
#pragma unroll
            for (int nt = 0; nt < 8; ++nt)
                acc = fmaf(fmaxf(c[mt][nt][r], 0.0f), w2v[nt], acc);
            part[mt][r] = acc;
        }
#pragma unroll
    for (int mt = 0; mt < 2; ++mt)
#pragma unroll
        for (int r = 0; r < 8; ++r) {
            float v = part[mt][r];
            v += __shfl_xor(v, 1, 32);
            v += __shfl_xor(v, 2, 32);
            v += __shfl_xor(v, 4, 32);
            v += __shfl_xor(v, 8, 32);
            part[mt][r] = v;
        }

    // Lane 0 writes rows 0..7, lane 16 writes rows 8..15 of each M-tile.
    if (mrow == 0) {
        const float bb2 = b2[0];
#pragma unroll
        for (int mt = 0; mt < 2; ++mt) {
#pragma unroll
            for (int r = 0; r < 8; ++r) {
                int e = e0 + mt * 16 + hio + r;
                if (e < E) {
                    float logit = part[mt][r] + bb2;
                    float uu = u[e];
                    float g  = logf(uu + GEPS) - logf(1.0f - uu + GEPS);
                    float y  = 1.0f / (1.0f + expf(-(logit + g)));  // TEMP=1
                    float yh = (y > 0.5f) ? 1.0f : 0.0f;            // == y_st
                    out[e]         = logit;
                    out[E + e]     = y;
                    out[2 * E + e] = yh;
                }
            }
        }
    }
}

extern "C" void kernel_launch(void* const* d_in, const int* in_sizes, int n_in,
                              void* d_out, int out_size, void* d_ws, size_t ws_size,
                              hipStream_t stream) {
    const float* x   = (const float*)d_in[0];
    const int*   ei  = (const int*)  d_in[1];
    const int*   et  = (const int*)  d_in[2];
    const float* u   = (const float*)d_in[3];
    const float* emb = (const float*)d_in[4];
    const float* W1  = (const float*)d_in[5];
    const float* b1  = (const float*)d_in[6];
    const float* W2  = (const float*)d_in[7];
    const float* b2  = (const float*)d_in[8];
    float* out = (float*)d_out;
    const int E  = in_sizes[1] / 2;
    const int NX = in_sizes[0];           // N_NODES*HIDDEN
    const int NE = in_sizes[4];           // N_ETYPES*ETYPE_DIM

    _Float16* bfrag = (_Float16*)d_ws;    // [0, 73728)
    const int prep_elems = NKB * NNT * 512;
    pgx_prep_w1<<<(prep_elems + 255) / 256, 256, 0, stream>>>(W1, bfrag);

    const int nwaves = (E + 31) / 32;
    const int blocks = (nwaves + 7) / 8;  // 8 waves (256 threads) per block

    const size_t need = (size_t)BFRAG_BYTES + (size_t)NX * 2 + (size_t)NE * 2;
    if (ws_size >= need) {
        // Pre-convert the gather tables to f16: halves gather bytes and
        // removes every v_cvt from the WMMA hot loop.
        _Float16* x16 = (_Float16*)((char*)d_ws + BFRAG_BYTES);
        _Float16* e16 = x16 + NX;
        pgx_cvt_f16<<<(NX + 255) / 256, 256, 0, stream>>>(x, x16, NX);
        pgx_cvt_f16<<<(NE + 255) / 256, 256, 0, stream>>>(emb, e16, NE);
        pgx_main<_Float16><<<blocks, 256, 0, stream>>>(
            x16, ei, et, u, e16, bfrag, b1, W2, b2, out, E);
    } else {
        // Fallback: gather fp32 and convert in-loop.
        pgx_main<float><<<blocks, 256, 0, stream>>>(
            x, ei, et, u, emb, bfrag, b1, W2, b2, out, E);
    }
}
